// ConvMESN_78125455114547
// MI455X (gfx1250) — matrix operand order repaired and checked by reading, not compiled, not run on hardware
//
#include <hip/hip_runtime.h>
#include <math.h>

// ---------------- constants (match reference) ----------------
#define NB      256      // batch
#define T_LEN   168
#define HID     400
#define KP      416      // HID padded to 13*32 for bf16 WMMA K-loop
#define FC_CH   48
#define N_OUT   24
#define N_SKIP  4
#define FT      3

typedef __attribute__((ext_vector_type(16))) __bf16 v16bf;
typedef __attribute__((ext_vector_type(8)))  float  v8f;

// ---------------- fragment loaders (bf16 16x16x32 WMMA) ----------------
// A 16x32: lane l holds row (l&15); K elems: K0+16h+0..15
__device__ __forceinline__ v16bf load_a_frag(const __bf16* __restrict__ row, int K0, int half) {
    const __bf16* p0 = row + K0 + half * 16;
    v16bf a;
#pragma unroll
    for (int e = 0; e < 16; ++e) a[e] = p0[e];
    return a;
}
// B 32x16 stored transposed [n][k]: lane l holds col (l&15); K elems: K0+16h+0..15
__device__ __forceinline__ v16bf load_b_frag(const __bf16* __restrict__ ncol, int K0, int half) {
    const __bf16* p = ncol + K0 + half * 16;
    v16bf b;
#pragma unroll
    for (int e = 0; e < 16; ++e) b[e] = p[e];
    return b;
}

// ---------------- utility kernels ----------------
__global__ void k_zero(unsigned* __restrict__ p, long long n) {
    long long i = (long long)blockIdx.x * blockDim.x + threadIdx.x;
    long long st = (long long)gridDim.x * blockDim.x;
    for (; i < n; i += st) p[i] = 0u;
}

// wres (4,400,400) f32 -> out[i][n][k] bf16, k padded to 416 with zeros (B transposed)
__global__ void k_cvt_wres(const float* __restrict__ w, __bf16* __restrict__ out) {
    long long idx = (long long)blockIdx.x * blockDim.x + threadIdx.x;
    long long total = (long long)N_SKIP * HID * KP;
    if (idx >= total) return;
    int k = (int)(idx % KP);
    long long r = idx / KP;
    int n = (int)(r % HID);
    int i = (int)(r / HID);
    float v = (k < HID) ? w[(long long)i * HID * HID + (long long)k * HID + n] : 0.f;
    out[idx] = (__bf16)v;
}

// conv weight (48,1,h,400) f32 -> out[c][dt][k] bf16, k padded to 416 (channel-major = B transposed)
__global__ void k_cvt_convw(const float* __restrict__ w, __bf16* __restrict__ out, int h) {
    long long idx = (long long)blockIdx.x * blockDim.x + threadIdx.x;
    long long total = (long long)FC_CH * h * KP;
    if (idx >= total) return;
    int k = (int)(idx % KP);
    long long r = idx / KP;
    int dt = (int)(r % h);
    int c  = (int)(r / h);
    float v = (k < HID) ? w[((long long)c * h + dt) * HID + k] : 0.f;
    out[idx] = (__bf16)v;
}

// ---------------- echo recurrence: one launch per (skip, t) ----------------
// xn = 0.5*xi + 0.5*tanh(x[m,t]*win[n] + (xi @ wres)[m,n]); ping-pong state buffers.
// Grid: 50 blocks * 256 thr = 400 waves = 16 m_tiles x 25 n_tiles.
__global__ void __launch_bounds__(256)
k_echo_step(const __bf16* __restrict__ sin_bf,   // [256][416] bf16 state (read)
            const float*  __restrict__ sin_f,    // [256][400] f32 state (read)
            __bf16*       __restrict__ sout_bf,  // [256][416] (write)
            float*        __restrict__ sout_f,   // [256][400] (write)
            const __bf16* __restrict__ wres_bt,  // [400][416] transposed, padded
            const float*  __restrict__ win_row,  // [400]
            const float*  __restrict__ x,        // batch_x: [m*T + t]
            __bf16*       __restrict__ echo,     // [(m*T + t)*416 + n]
            int t) {
    int tid  = blockIdx.x * blockDim.x + threadIdx.x;
    int wid  = tid >> 5;
    int lane = tid & 31;
    int half = lane >> 4, lmod = lane & 15;
    int m_tile = wid / 25, n_tile = wid % 25;
    int m0 = m_tile * 16, n0 = n_tile * 16;

    const __bf16* arow = sin_bf  + (long long)(m0 + lmod) * KP;
    const __bf16* brow = wres_bt + (long long)(n0 + lmod) * KP;

    v8f acc = {0.f, 0.f, 0.f, 0.f, 0.f, 0.f, 0.f, 0.f};
#pragma unroll
    for (int kb = 0; kb < 13; ++kb) {
        v16bf a = load_a_frag(arow, kb * 32, half);
        v16bf b = load_b_frag(brow, kb * 32, half);
        acc = __builtin_amdgcn_wmma_f32_16x16x32_bf16(false, a, false, b,
                                                      (short)0, acc, false, false);
    }

    int n = n0 + lmod;
    float wn = win_row[n];
#pragma unroll
    for (int r = 0; r < 8; ++r) {
        int m = m0 + r + half * 8;
        float u   = x[(long long)m * T_LEN + t];
        float val = acc[r] + u * wn;
        float xup = tanhf(val);
        float xi  = sin_f[(long long)m * HID + n];
        float xn  = 0.5f * (xi + xup);   // LEAKY = 0.5
        sout_f[(long long)m * HID + n] = xn;
        sout_bf[(long long)m * KP + n] = (__bf16)xn;
        echo[((long long)m * T_LEN + t) * KP + n] = (__bf16)xn;
    }
}

// ---------------- conv-as-GEMM + ReLU + max-pool (bitwise atomicMax) ----------------
// Register-blocked: each wave owns a 32-batch x 48-channel tile at one position p
// (2 m-tiles x 3 n-tiles = 6 f32 WMMA accumulators). Per kb step: 2 A frags + 3 B
// frags -> 6 WMMAs (1.67 b128 loads per wmma instead of 4).
// Grid: P blocks * 8 waves; wave w = m-block (32 rows), blockIdx.x = p.
__global__ void __launch_bounds__(256)
k_conv_max(const __bf16* __restrict__ echo,   // [m][t][416]
           const __bf16* __restrict__ wbt,    // [c][dt][416]
           const float*  __restrict__ bias,   // [48]
           unsigned*     __restrict__ pool,   // [256*48] f32 bits, pre-zeroed
           int h) {
    int lane = threadIdx.x & 31;
    int half = lane >> 4, lmod = lane & 15;
    int p    = blockIdx.x;
    int m0   = (threadIdx.x >> 5) * 32;      // 8 waves -> m blocks of 32

    const __bf16* mrow0 = echo + (long long)(m0 + lmod) * T_LEN * KP;
    const __bf16* mrow1 = echo + (long long)(m0 + 16 + lmod) * T_LEN * KP;
    const __bf16* crow0 = wbt + (long long)(0  + lmod) * h * KP;
    const __bf16* crow1 = wbt + (long long)(16 + lmod) * h * KP;
    const __bf16* crow2 = wbt + (long long)(32 + lmod) * h * KP;

    v8f acc[2][3];
#pragma unroll
    for (int mt = 0; mt < 2; ++mt)
#pragma unroll
        for (int ct = 0; ct < 3; ++ct)
            acc[mt][ct] = (v8f){0.f, 0.f, 0.f, 0.f, 0.f, 0.f, 0.f, 0.f};

    for (int dt = 0; dt < h; ++dt) {
        long long aoff = (long long)(p + dt) * KP;
        long long boff = (long long)dt * KP;
#pragma unroll
        for (int kb = 0; kb < 13; ++kb) {
            int K0 = kb * 32;
            v16bf a0 = load_a_frag(mrow0 + aoff, K0, half);
            v16bf a1 = load_a_frag(mrow1 + aoff, K0, half);
            v16bf b0 = load_b_frag(crow0 + boff, K0, half);
            v16bf b1 = load_b_frag(crow1 + boff, K0, half);
            v16bf b2 = load_b_frag(crow2 + boff, K0, half);
            acc[0][0] = __builtin_amdgcn_wmma_f32_16x16x32_bf16(false, a0, false, b0, (short)0, acc[0][0], false, false);
            acc[0][1] = __builtin_amdgcn_wmma_f32_16x16x32_bf16(false, a0, false, b1, (short)0, acc[0][1], false, false);
            acc[0][2] = __builtin_amdgcn_wmma_f32_16x16x32_bf16(false, a0, false, b2, (short)0, acc[0][2], false, false);
            acc[1][0] = __builtin_amdgcn_wmma_f32_16x16x32_bf16(false, a1, false, b0, (short)0, acc[1][0], false, false);
            acc[1][1] = __builtin_amdgcn_wmma_f32_16x16x32_bf16(false, a1, false, b1, (short)0, acc[1][1], false, false);
            acc[1][2] = __builtin_amdgcn_wmma_f32_16x16x32_bf16(false, a1, false, b2, (short)0, acc[1][2], false, false);
        }
    }

#pragma unroll
    for (int ct = 0; ct < 3; ++ct) {
        int c  = ct * 16 + lmod;
        float bb = bias[c];
#pragma unroll
        for (int mt = 0; mt < 2; ++mt) {
#pragma unroll
            for (int r = 0; r < 8; ++r) {
                int m = m0 + mt * 16 + r + half * 8;
                float v = fmaxf(acc[mt][ct][r] + bb, 0.f);   // ReLU => v >= 0
                atomicMax(&pool[(long long)m * FC_CH + c], __float_as_uint(v));
            }
        }
    }
}

// ---------------- final FC: (256,576) @ (576,24)^T + b ----------------
__global__ void k_fc(const float* __restrict__ pools,  // [12][256][48]
                     const float* __restrict__ fcw,    // [24][576]
                     const float* __restrict__ fcb,    // [24]
                     float* __restrict__ out) {
    int tid = blockIdx.x * blockDim.x + threadIdx.x;
    if (tid >= NB * N_OUT) return;
    int m = tid / N_OUT, o = tid % N_OUT;
    float s = fcb[o];
#pragma unroll 1
    for (int k = 0; k < N_SKIP * FT; ++k)
        for (int c = 0; c < FC_CH; ++c)
            s += pools[((long long)k * NB + m) * FC_CH + c] * fcw[(long long)o * 576 + k * FC_CH + c];
    out[(long long)m * N_OUT + o] = s;
}

// ---------------- host orchestration ----------------
extern "C" void kernel_launch(void* const* d_in, const int* in_sizes, int n_in,
                              void* d_out, int out_size, void* d_ws, size_t ws_size,
                              hipStream_t stream) {
    (void)in_sizes; (void)n_in; (void)out_size; (void)ws_size;
    // dict order: batch_x, win, wres, conv_ws[12], conv_bs[12], fc_w, fc_b
    const float* batch_x = (const float*)d_in[0];
    const float* win     = (const float*)d_in[1];   // (4,1,400)
    const float* wres    = (const float*)d_in[2];   // (4,400,400)
    const float* convw[12]; const float* convb[12];
    for (int k = 0; k < 12; ++k) { convw[k] = (const float*)d_in[3 + k];
                                   convb[k] = (const float*)d_in[15 + k]; }
    const float* fcw = (const float*)d_in[27];
    const float* fcb = (const float*)d_in[28];
    float* out = (float*)d_out;

    static const int skips[4] = {1, 3, 9, 27};
    static const int hs[12]   = {1, 2, 3, 3, 6, 9, 9, 18, 27, 27, 54, 81};
    static const int slot_base[4] = {0, 2, 8, 26};  // (slot,phase)-buffer base per skip

    char* ws = (char*)d_ws;
    size_t off = 0;
    auto alloc = [&](size_t bytes) { size_t o = off; off = (off + bytes + 255) & ~(size_t)255; return o; };

    __bf16* wres_bt = (__bf16*)(ws + alloc((size_t)N_SKIP * HID * KP * 2));
    __bf16* wbt[12];
    for (int k = 0; k < 12; ++k)
        wbt[k] = (__bf16*)(ws + alloc((size_t)FC_CH * hs[k] * KP * 2));

    size_t zstart = off;
    __bf16* state_bf = (__bf16*)(ws + alloc((size_t)80 * NB * KP * 2));   // 40 slots x 2 phases
    float*  state_f  = (float*) (ws + alloc((size_t)80 * NB * HID * 4));
    __bf16* echo     = (__bf16*)(ws + alloc((size_t)NB * T_LEN * KP * 2));
    unsigned* pools  = (unsigned*)(ws + alloc((size_t)12 * NB * FC_CH * 4));
    size_t zbytes = off - zstart;

    // 1) zero states (ping-pong init + pad cols), echo pad, pools
    k_zero<<<2048, 256, 0, stream>>>((unsigned*)(ws + zstart), (long long)(zbytes / 4));

    // 2) weight conversions to bf16 (transposed, K padded to 416)
    {
        long long tot = (long long)N_SKIP * HID * KP;
        k_cvt_wres<<<(unsigned)((tot + 255) / 256), 256, 0, stream>>>(wres, wres_bt);
    }
    for (int k = 0; k < 12; ++k) {
        long long tot = (long long)FC_CH * hs[k] * KP;
        k_cvt_convw<<<(unsigned)((tot + 255) / 256), 256, 0, stream>>>(convw[k], wbt[k], hs[k]);
    }

    // 3) per skip: sequential echo recurrence, then the 3 conv+pool GEMMs
    for (int i = 0; i < N_SKIP; ++i) {
        int s = skips[i];
        for (int t = 0; t < T_LEN; ++t) {
            int idx = t % s;
            int q   = t / s;
            int pr  = q & 1;
            size_t bin  = (size_t)(slot_base[i] + idx * 2 + pr);
            size_t bout = (size_t)(slot_base[i] + idx * 2 + (1 - pr));
            k_echo_step<<<50, 256, 0, stream>>>(
                state_bf + bin  * NB * KP,  state_f + bin  * NB * HID,
                state_bf + bout * NB * KP,  state_f + bout * NB * HID,
                wres_bt + (size_t)i * HID * KP,
                win + (size_t)i * HID,
                batch_x, echo, t);
        }
        for (int j = 0; j < FT; ++j) {
            int k = i * FT + j;
            int h = s * (j + 1);
            int P = T_LEN - h + 1;
            k_conv_max<<<P, 256, 0, stream>>>(
                echo, wbt[k], convb[k], pools + (size_t)k * NB * FC_CH, h);
        }
    }

    // 4) final FC
    k_fc<<<(NB * N_OUT + 255) / 256, 256, 0, stream>>>((const float*)pools, fcw, fcb, out);
}
